// TestTimeSuSXShiftTuning_61924838474033
// MI455X (gfx1250) — matrix operand order repaired and checked by reading, not compiled, not run on hardware
//
#include <hip/hip_runtime.h>
#include <stdint.h>

#define N_IMG 400
#define N_PAD 416          // 400 padded to multiple of 32 for 2x2 WMMA blocking
#define D_DIM 512
#define C_CLS 1000
#define C_PAD 1024
#define M_SUP 16000

typedef __attribute__((ext_vector_type(16))) __bf16        v16bf;
typedef __attribute__((ext_vector_type(8)))  float         v8f;
typedef __attribute__((ext_vector_type(4)))  unsigned int  u32x4;

struct FragBits { u32x4 a, b; };

__device__ __forceinline__ uint16_t f2bf(float f) {
    uint32_t u = __float_as_uint(f);
    uint32_t r = u + 0x7FFFu + ((u >> 16) & 1u);   // round-to-nearest-even
    return (uint16_t)(r >> 16);
}

__device__ __forceinline__ v16bf bcast(const FragBits& fb) {
    return __builtin_bit_cast(v16bf, fb);
}

// block-wide reduction, blockDim.x == 256. mode: 0=sum 1=max 2=min
__device__ __forceinline__ float blkRed(float v, int mode) {
    __shared__ float red[256];
    int t = threadIdx.x;
    __syncthreads();
    red[t] = v;
    __syncthreads();
    for (int s = 128; s > 0; s >>= 1) {
        if (t < s) {
            float a = red[t], b = red[t + s];
            red[t] = (mode == 0) ? (a + b) : (mode == 1 ? fmaxf(a, b) : fminf(a, b));
        }
        __syncthreads();
    }
    return red[0];
}

// ---- L2-normalize rows of `in` [rows x cols] fp32 -> bf16; rows >= validRows zero-padded
__global__ void k_norm_rows_bf16(const float* __restrict__ in, uint16_t* __restrict__ out,
                                 int validRows, int cols) {
    int r = blockIdx.x;
    if (r >= validRows) {
        for (int i = threadIdx.x; i < cols; i += 256) out[(size_t)r * cols + i] = 0;
        return;
    }
    const float* row = in + (size_t)r * cols;
    float ss = 0.f;
    for (int i = threadIdx.x; i < cols; i += 256) { float x = row[i]; ss += x * x; }
    ss = blkRed(ss, 0);
    float inv = 1.f / fmaxf(sqrtf(ss), 1e-12f);
    for (int i = threadIdx.x; i < cols; i += 256)
        out[(size_t)r * cols + i] = f2bf(row[i] * inv);
}

// ---- transpose support_feats [D x M] fp32 -> supT [M x D] bf16 (32x32 LDS tiles)
__global__ void k_transpose_cvt(const float* __restrict__ S, uint16_t* __restrict__ T) {
    __shared__ float t[32][33];
    int tx = threadIdx.x, ty = threadIdx.y;           // block (32, 8)
    int mBase = blockIdx.x * 32, dBase = blockIdx.y * 32;
    #pragma unroll
    for (int j = 0; j < 32; j += 8)
        t[ty + j][tx] = S[(size_t)(dBase + ty + j) * M_SUP + (mBase + tx)];
    __syncthreads();
    #pragma unroll
    for (int j = 0; j < 32; j += 8)
        T[(size_t)(mBase + ty + j) * D_DIM + (dBase + tx)] = f2bf(t[tx][ty + j]);
}

// ---- bf16 WMMA GEMM, 2x2 register blocking (32x32 out per wave, 4 waves/block):
//   C[Mrows x Ncols] = alpha * A[mPad x K] * B[Ncols x K]^T (+ rowbias[row])
// A must be padded (zero rows) up to mPad = roundup(Mrows,32). 4 b128 loads / WMMA
// drops to 2, halving L2 fragment traffic and keeping 4 independent WMMAs in flight.
__global__ void k_gemm_bf16(const uint16_t* __restrict__ A, const uint16_t* __restrict__ B,
                            float* __restrict__ C, int Mrows, int Ncols, int K,
                            float alpha, const float* __restrict__ rowbias) {
    int wave = threadIdx.x >> 5, lane = threadIdx.x & 31;
    int mPad = (Mrows + 31) & ~31;
    int tm2 = blockIdx.y * 4 + wave;                  // 32-row tile index
    if (tm2 * 32 >= mPad) return;
    int tn2 = blockIdx.x;                             // 32-col tile index
    int g = lane >> 4, l = lane & 15;

    const uint16_t* arow0 = A + (size_t)(tm2 * 32 + l) * K;
    const uint16_t* arow1 = A + (size_t)(tm2 * 32 + 16 + l) * K;
    const uint16_t* brow0 = B + (size_t)(tn2 * 32 + l) * K;
    const uint16_t* brow1 = B + (size_t)(tn2 * 32 + 16 + l) * K;

    v8f acc00 = {}, acc01 = {}, acc10 = {}, acc11 = {};
    for (int k0 = 0; k0 < K; k0 += 32) {
        FragBits fa0, fa1, fb0, fb1;
        // A 16x32 bf16 fragment: lane group g holds K halves {k0+8g..+7} and {k0+16+8g..+7}
        fa0.a = *(const u32x4*)(arow0 + k0 + 8 * g);
        fa0.b = *(const u32x4*)(arow0 + k0 + 16 + 8 * g);
        fa1.a = *(const u32x4*)(arow1 + k0 + 8 * g);
        fa1.b = *(const u32x4*)(arow1 + k0 + 16 + 8 * g);
        // B 32x16 bf16 fragment: lane group g holds K halves {k0+8g..+7} and {k0+16+8g..+7}
        fb0.a = *(const u32x4*)(brow0 + k0 + 8 * g);
        fb0.b = *(const u32x4*)(brow0 + k0 + 16 + 8 * g);
        fb1.a = *(const u32x4*)(brow1 + k0 + 8 * g);
        fb1.b = *(const u32x4*)(brow1 + k0 + 16 + 8 * g);
        v16bf a0 = bcast(fa0), a1 = bcast(fa1), b0 = bcast(fb0), b1 = bcast(fb1);
        acc00 = __builtin_amdgcn_wmma_f32_16x16x32_bf16(false, a0, false, b0, (short)0, acc00, false, false);
        acc01 = __builtin_amdgcn_wmma_f32_16x16x32_bf16(false, a0, false, b1, (short)0, acc01, false, false);
        acc10 = __builtin_amdgcn_wmma_f32_16x16x32_bf16(false, a1, false, b0, (short)0, acc10, false, false);
        acc11 = __builtin_amdgcn_wmma_f32_16x16x32_bf16(false, a1, false, b1, (short)0, acc11, false, false);
    }

    auto store_tile = [&](const v8f& acc, int mo, int no) {
        #pragma unroll
        for (int r = 0; r < 8; ++r) {
            int orow = tm2 * 32 + mo + r + 8 * g;     // D layout: VGPR r -> M=r (+8 for lanes 16-31)
            if (orow < Mrows) {
                float v = alpha * acc[r];
                if (rowbias) v += rowbias[orow];
                C[(size_t)orow * Ncols + tn2 * 32 + no + l] = v;
            }
        }
    };
    store_tile(acc00, 0, 0);
    store_tile(acc01, 0, 16);
    store_tile(acc10, 16, 0);
    store_tile(acc11, 16, 16);
}

// ---- log-softmax over first C_CLS cols of fp32 [.. x C_PAD] row -> bf16, pads zeroed
__global__ void k_logsoftmax_bf16(const float* __restrict__ logits, uint16_t* __restrict__ out) {
    int m = blockIdx.x;
    const float* row = logits + (size_t)m * C_PAD;
    float mx = -__builtin_inff();
    for (int i = threadIdx.x; i < C_CLS; i += 256) mx = fmaxf(mx, row[i]);
    mx = blkRed(mx, 1);
    float s = 0.f;
    for (int i = threadIdx.x; i < C_CLS; i += 256) s += __expf(row[i] - mx);
    s = blkRed(s, 0);
    float lz = mx + __logf(s);
    for (int i = threadIdx.x; i < C_CLS; i += 256)
        out[(size_t)m * C_PAD + i] = f2bf(row[i] - lz);
    for (int i = C_CLS + threadIdx.x; i < C_PAD; i += 256)
        out[(size_t)m * C_PAD + i] = 0;
}

// ---- test softmax: p (bf16, pads 0, rows >= N_IMG zeroed) and entropy[n] = sum p*log p
__global__ void k_softmax_test(const float* __restrict__ logits, uint16_t* __restrict__ pout,
                               float* __restrict__ entropy) {
    int n = blockIdx.x;
    if (n >= N_IMG) {                                  // zero pad rows (grid = N_PAD)
        for (int i = threadIdx.x; i < C_PAD; i += 256) pout[(size_t)n * C_PAD + i] = 0;
        return;
    }
    const float* row = logits + (size_t)n * C_PAD;
    float mx = -__builtin_inff();
    for (int i = threadIdx.x; i < C_CLS; i += 256) mx = fmaxf(mx, row[i]);
    mx = blkRed(mx, 1);
    float s = 0.f;
    for (int i = threadIdx.x; i < C_CLS; i += 256) s += __expf(row[i] - mx);
    s = blkRed(s, 0);
    float lz = mx + __logf(s);
    float ent = 0.f;
    for (int i = threadIdx.x; i < C_CLS; i += 256) {
        float lp = row[i] - lz;
        float p = __expf(lp);
        pout[(size_t)n * C_PAD + i] = f2bf(p);
        ent += p * lp;
    }
    ent = blkRed(ent, 0);
    if (threadIdx.x == 0) entropy[n] = ent;
    for (int i = C_CLS + threadIdx.x; i < C_PAD; i += 256)
        pout[(size_t)n * C_PAD + i] = 0;
}

// ---- stage 1 global min/max of kl and nk buffers
__global__ void k_minmax_part(const float* __restrict__ kl, const float* __restrict__ nk,
                              float* __restrict__ part) {
    size_t total = (size_t)N_IMG * M_SUP;
    float kmin = __builtin_inff(), kmax = -__builtin_inff();
    float nmin = __builtin_inff(), nmax = -__builtin_inff();
    for (size_t i = (size_t)blockIdx.x * 256 + threadIdx.x; i < total; i += (size_t)gridDim.x * 256) {
        float k = kl[i], v = nk[i];
        kmin = fminf(kmin, k); kmax = fmaxf(kmax, k);
        nmin = fminf(nmin, v); nmax = fmaxf(nmax, v);
    }
    kmin = blkRed(kmin, 2); kmax = blkRed(kmax, 1);
    nmin = blkRed(nmin, 2); nmax = blkRed(nmax, 1);
    if (threadIdx.x == 0) {
        part[blockIdx.x * 4 + 0] = kmin; part[blockIdx.x * 4 + 1] = kmax;
        part[blockIdx.x * 4 + 2] = nmin; part[blockIdx.x * 4 + 3] = nmax;
    }
}

__global__ void k_minmax_final(const float* __restrict__ part, float* __restrict__ sc) {
    int t = threadIdx.x;    // 256 partials
    float kmin = part[t * 4 + 0], kmax = part[t * 4 + 1];
    float nmin = part[t * 4 + 2], nmax = part[t * 4 + 3];
    kmin = blkRed(kmin, 2); kmax = blkRed(kmax, 1);
    nmin = blkRed(nmin, 2); nmax = blkRed(nmax, 1);
    if (t == 0) { sc[0] = kmin; sc[1] = kmax; sc[2] = nmin; sc[3] = nmax; }
}

// ---- per-image label scatter (LDS ds_add_f32) + fused final combine
__global__ void k_final(const float* __restrict__ kl, const float* __restrict__ nk,
                        const int* __restrict__ labels, const float* __restrict__ sc,
                        const float* __restrict__ test_logits, float* __restrict__ out) {
    int n = blockIdx.x;
    __shared__ float s[2 * C_PAD];                    // [0,C_PAD): kl bins, [C_PAD,..): cache bins
    for (int i = threadIdx.x; i < 2 * C_PAD; i += 256) s[i] = 0.f;
    __syncthreads();
    float klmin = sc[0], klmax = sc[1], nkmin = sc[2], nkmax = sc[3];
    float scale = (nkmax - nkmin) / (klmax - klmin);
    for (int m = threadIdx.x; m < M_SUP; m += 256) {
        float kv = kl[(size_t)n * M_SUP + m];
        float nv = nk[(size_t)n * M_SUP + m];
        float aff = -((kv - klmin) * scale + nkmin);  // -scale(kl -> nk range)
        float w = __expf(5.5f * (nv - 1.f));          // exp(-(BETA - BETA*nk))
        int c = labels[m];
        atomicAdd(&s[c], aff);
        atomicAdd(&s[C_PAD + c], w);
    }
    __syncthreads();
    for (int c = threadIdx.x; c < C_CLS; c += 256)
        out[(size_t)n * C_CLS + c] =
            50.f * test_logits[(size_t)n * C_PAD + c]   // 100 * (img @ tf) with logits = x/TEMP
            + 5.0f * s[c]                               // GAMMA * kl_logits
            + 1.0f * s[C_PAD + c];                      // ALPHA * cache_logits
}

extern "C" void kernel_launch(void* const* d_in, const int* in_sizes, int n_in,
                              void* d_out, int out_size, void* d_ws, size_t ws_size,
                              hipStream_t stream) {
    (void)in_sizes; (void)n_in; (void)out_size; (void)ws_size;
    const float* img = (const float*)d_in[0];   // [400, 512]
    const float* txt = (const float*)d_in[1];   // [1000, 512]
    const float* sup = (const float*)d_in[2];   // [512, 16000] (already col-normalized)
    const int*   lbl = (const int*)d_in[3];     // [16000]
    float* out = (float*)d_out;                 // [400, 1000]

    char* w = (char*)d_ws; size_t off = 0;
    auto take = [&](size_t bytes) -> char* {
        char* p = w + off; off += (bytes + 255) & ~(size_t)255; return p;
    };
    uint16_t* tf_bf    = (uint16_t*)take((size_t)C_PAD * D_DIM * 2);
    uint16_t* img_bf   = (uint16_t*)take((size_t)N_PAD * D_DIM * 2);   // zero-padded rows
    uint16_t* supT_bf  = (uint16_t*)take((size_t)M_SUP * D_DIM * 2);
    float*    train_lg = (float*)   take((size_t)M_SUP * C_PAD * 4);
    uint16_t* logq_bf  = (uint16_t*)take((size_t)M_SUP * C_PAD * 2);
    float*    test_lg  = (float*)   take((size_t)N_IMG * C_PAD * 4);
    uint16_t* p_bf     = (uint16_t*)take((size_t)N_PAD * C_PAD * 2);   // zero-padded rows
    float*    entropy  = (float*)   take((size_t)N_IMG * 4);
    float*    klb      = (float*)   take((size_t)N_IMG * M_SUP * 4);
    float*    nkb      = (float*)   take((size_t)N_IMG * M_SUP * 4);
    float*    part     = (float*)   take(256 * 4 * 4);
    float*    scal     = (float*)   take(4 * 4);

    // 1) normalize + bf16 convert (img padded to N_PAD zero rows)
    k_norm_rows_bf16<<<C_PAD, 256, 0, stream>>>(txt, tf_bf, C_CLS, D_DIM);
    k_norm_rows_bf16<<<N_PAD, 256, 0, stream>>>(img, img_bf, N_IMG, D_DIM);
    // 2) support [D,M] -> [M,D] bf16
    k_transpose_cvt<<<dim3(M_SUP / 32, D_DIM / 32), dim3(32, 8), 0, stream>>>(sup, supT_bf);
    // 3) train_logits [M, C_PAD] = (supT @ tf^T) / TEMP      (500 row-tiles / 4 waves)
    k_gemm_bf16<<<dim3(C_PAD / 32, M_SUP / 128), 128, 0, stream>>>(
        supT_bf, tf_bf, train_lg, M_SUP, C_PAD, D_DIM, 2.0f, nullptr);
    // 4) test_logits [N, C_PAD] = (img @ tf^T) / TEMP        (13 row-tiles -> 4 blocks)
    k_gemm_bf16<<<dim3(C_PAD / 32, 4), 128, 0, stream>>>(
        img_bf, tf_bf, test_lg, N_IMG, C_PAD, D_DIM, 2.0f, nullptr);
    // 5) softmaxes
    k_logsoftmax_bf16<<<M_SUP, 256, 0, stream>>>(train_lg, logq_bf);
    k_softmax_test<<<N_PAD, 256, 0, stream>>>(test_lg, p_bf, entropy);
    // 6) kl[n,m] = entropy[n] - p @ logq^T   (rowbias epilogue, alpha = -1)
    k_gemm_bf16<<<dim3(M_SUP / 32, 4), 128, 0, stream>>>(
        p_bf, logq_bf, klb, N_IMG, M_SUP, C_PAD, -1.0f, entropy);
    // 7) new_knowledge[n,m] = img @ sup
    k_gemm_bf16<<<dim3(M_SUP / 32, 4), 128, 0, stream>>>(
        img_bf, supT_bf, nkb, N_IMG, M_SUP, D_DIM, 1.0f, nullptr);
    // 8) global min/max of kl and nk
    k_minmax_part<<<256, 256, 0, stream>>>(klb, nkb, part);
    k_minmax_final<<<1, 256, 0, stream>>>(part, scal);
    // 9) scatter-by-label + fused final combine
    k_final<<<N_IMG, 256, 0, stream>>>(klb, nkb, lbl, scal, test_lg, out);
}
